// CartesianMACE_10144712753197
// MI455X (gfx1250) — compile-verified, matrix operand order checked by reading, not run on hardware
//
#include <hip/hip_runtime.h>
#include <hip/hip_bf16.h>

#define NNODES 10000
#define NEDGES 256000
#define H      128
#define PDIM   64
#define NBESS  8
#define NSPEC  10
#define RCUT   5.0f
#define INV_AVG (1.0f/25.6f)

typedef __attribute__((ext_vector_type(16))) _Float16 v16h;
typedef __attribute__((ext_vector_type(8)))  _Float16 v8h;
typedef __attribute__((ext_vector_type(8)))  float    v8f;

// ---------------------------------------------------------------------------
// WMMA helpers (wave32, v_wmma_f32_16x16x32_f16)
//
// A (16xK f16, row-major in LDS): lane group g=lane>>4, m=lane&15.
//   lane's halves = A[m][kt*32 + 8g + 0..7] ++ A[m][kt*32 + 16 + 8g + 0..7]
//   -> two aligned 16-byte LDS loads.
//
// B (KxN f16): pre-swizzled at LDS-fill time into fragment order so each
//   lane reads 32 contiguous bytes per (ktile,ntile): packed index
//   ((nt*KT + kt)*32 + lane)*16 + h, where h=2v+odd maps to
//   k = kt*32 + k0(v,g) + odd, col = nt*16 + n.
// ---------------------------------------------------------------------------
__device__ __forceinline__ constexpr int k0_of(int v, int g) {
  return (v < 4) ? (2*v + 8*g) : (16 + 2*(v-4) + 8*g);
}

// Convert f32 global weights -> f16 LDS in B-fragment-packed order.
template<int K, int N, int VALIDK>
__device__ __forceinline__ void fill_packed(_Float16* __restrict__ dst,
                                            const float* __restrict__ src,
                                            int tid, int nthr) {
  constexpr int KT = K / 32;
  for (int idx = tid; idx < K*N; idx += nthr) {
    const int h    = idx & 15;
    const int lane = (idx >> 4) & 31;
    const int t    = idx >> 9;
    const int kt   = t % KT;
    const int nt   = t / KT;
    const int g = (lane >> 4) & 1, n = lane & 15;
    const int v = h >> 1, odd = h & 1;
    const int k   = kt*32 + k0_of(v, g) + odd;
    const int col = nt*16 + n;
    dst[idx] = (_Float16)((k < VALIDK) ? src[k*N + col] : 0.0f);
  }
}

// 16xK @ KxN -> f16 LDS out (row-major), optional fast SiLU on f32 accum.
template<bool SILU, int K, int N, int LDA, int LDD>
__device__ __forceinline__ void wmma_layer(const _Float16* __restrict__ act,
                                           const _Float16* __restrict__ Wpk,
                                           _Float16* __restrict__ dst, int lane) {
  constexpr int KT = K / 32;
  const int g = (lane >> 4) & 1, n = lane & 15, m = lane & 15;
  for (int nt = 0; nt < N/16; ++nt) {
    v8f c = {};
#pragma unroll
    for (int kt = 0; kt < KT; ++kt) {
      const _Float16* arow = act + m*LDA + kt*32 + 8*g;
      const v8h alo = *(const v8h*)(arow);
      const v8h ahi = *(const v8h*)(arow + 16);
      const v16h a = __builtin_shufflevector(alo, ahi,
                       0,1,2,3,4,5,6,7,8,9,10,11,12,13,14,15);
      const v16h b = *(const v16h*)(Wpk + (((nt*KT + kt)*32 + lane) << 4));
      c = __builtin_amdgcn_wmma_f32_16x16x32_f16(false, a, false, b, (short)0, c,
                                                 false, false);
    }
#pragma unroll
    for (int r = 0; r < 8; ++r) {
      float x = c[r];
      if (SILU) x = x * __builtin_amdgcn_rcpf(1.0f + __expf(-x));  // fast SiLU
      dst[(r + 8*g)*LDD + nt*16 + n] = (_Float16)x;
    }
  }
}

// ---------------------------------------------------------------------------
// Small utility kernels
// ---------------------------------------------------------------------------
__global__ void zero_kernel(float* __restrict__ p, int n) {
  for (int i = blockIdx.x*blockDim.x + threadIdx.x; i < n; i += gridDim.x*blockDim.x)
    p[i] = 0.0f;
}

__global__ void species_kernel(const float* __restrict__ attrs, int* __restrict__ spec) {
  int n = blockIdx.x*blockDim.x + threadIdx.x;
  if (n >= NNODES) return;
  int s = 0; float best = attrs[n*NSPEC];
  for (int k = 1; k < NSPEC; ++k) {
    float v = attrs[n*NSPEC + k];
    if (v > best) { best = v; s = k; }
  }
  spec[n] = s;
}

// Per-species tables (one-hot node_attrs shortcut):
// h0_s = W_embed[s] @ Wup0_1 ; sc0_s[h] = sum_c Wsc1[s,c,h]*W_embed[s,c]
__global__ void tables_kernel(const float* __restrict__ W_embed,
                              const float* __restrict__ Wup0_1,
                              const float* __restrict__ Wsc1,
                              float* __restrict__ h0tab, float* __restrict__ sc0tab) {
  const int s = blockIdx.x;   // NSPEC
  const int h = threadIdx.x;  // H
  float h0 = 0.0f, sc = 0.0f;
  for (int c = 0; c < H; ++c) {
    const float f = W_embed[s*H + c];
    h0 = fmaf(f, Wup0_1[c*H + h], h0);
    sc = fmaf(Wsc1[(s*H + c)*H + h], f, sc);
  }
  h0tab[s*H + h]  = h0;
  sc0tab[s*H + h] = sc;
}

// Per-edge: unit vector Y1 and 8 Bessel radial features with polynomial cutoff.
__global__ void edge_geom_kernel(const float* __restrict__ pos,
                                 const float* __restrict__ shifts,
                                 const int* __restrict__ idx_i,
                                 const int* __restrict__ idx_j,
                                 float* __restrict__ Y1, float* __restrict__ feats) {
  int e = blockIdx.x*blockDim.x + threadIdx.x;
  if (e >= NEDGES) return;
  const int i = idx_i[e], j = idx_j[e];
  const float vx = pos[i*3+0] - pos[j*3+0] - shifts[e*3+0];
  const float vy = pos[i*3+1] - pos[j*3+1] - shifts[e*3+1];
  const float vz = pos[i*3+2] - pos[j*3+2] - shifts[e*3+2];
  float r = sqrtf(vx*vx + vy*vy + vz*vz);
  r = fmaxf(r, 1e-9f);
  const float inv_r = 1.0f / r;
  Y1[e*3+0] = vx*inv_r; Y1[e*3+1] = vy*inv_r; Y1[e*3+2] = vz*inv_r;
  const float u = r / RCUT;
  float env = 0.0f;
  if (u < 1.0f) {  // p=5: 1 - 21 u^5 + 35 u^6 - 15 u^7
    const float u2 = u*u, u5 = u2*u2*u;
    env = 1.0f - 21.0f*u5 + 35.0f*u5*u - 15.0f*u5*u2;
  }
  const float pref = sqrtf(2.0f/RCUT) * inv_r * env;
  const float w = 3.14159265358979323846f * r / RCUT;
#pragma unroll
  for (int k = 0; k < NBESS; ++k)
    feats[e*NBESS + k] = pref * __sinf((float)(k+1) * w);
}

// ---------------------------------------------------------------------------
// Edge kernel 1: fused radial-MLP (WMMA) -> messages -> atomic scatter
// Block = 4 waves; each wave owns a 16-edge tile. Weights packed f16 in LDS.
// ---------------------------------------------------------------------------
#define EWAVES 4
__global__ void __launch_bounds__(32*EWAVES)
edge_mlp1_kernel(const float* __restrict__ feats, const float* __restrict__ Y1,
                 const int* __restrict__ idx_i, const int* __restrict__ idx_j,
                 const int* __restrict__ spec, const float* __restrict__ h0tab_g,
                 const float* __restrict__ R0, const float* __restrict__ R1,
                 const float* __restrict__ R2, const float* __restrict__ R3,
                 float* __restrict__ a0, float* __restrict__ a1) {
  __shared__ alignas(32) _Float16 W1h[32*64];       // padded 8->32 rows
  __shared__ alignas(32) _Float16 W2h[64*64];
  __shared__ alignas(32) _Float16 W3h[64*64];
  __shared__ alignas(32) _Float16 W4h[64*256];
  __shared__ float    h0s[NSPEC*H];
  __shared__ alignas(16) _Float16 act1[EWAVES][16*32];
  __shared__ alignas(16) _Float16 act2[EWAVES][16*64];
  __shared__ alignas(16) _Float16 act3[EWAVES][16*64];
  __shared__ alignas(16) _Float16 wt  [EWAVES][16*256];

  const int tid = threadIdx.x, nthr = blockDim.x;
  fill_packed<32,  64, NBESS>(W1h, R0, tid, nthr);
  fill_packed<64,  64, 64  >(W2h, R1, tid, nthr);
  fill_packed<64,  64, 64  >(W3h, R2, tid, nthr);
  fill_packed<64, 256, 64  >(W4h, R3, tid, nthr);
  for (int i = tid; i < NSPEC*H; i += nthr) h0s[i] = h0tab_g[i];
  __syncthreads();

  const int lane = tid & 31, wave = tid >> 5;
  const int ntiles = NEDGES / 16;
  for (int tile = blockIdx.x*EWAVES + wave; tile < ntiles; tile += gridDim.x*EWAVES) {
    const int e0 = tile * 16;
    _Float16* A1p = act1[wave];
    _Float16* A2p = act2[wave];
    _Float16* A3p = act3[wave];
    _Float16* Wp  = wt[wave];
    for (int idx = lane; idx < 16*32; idx += 32) {
      const int m = idx >> 5, k = idx & 31;
      A1p[idx] = (_Float16)((k < NBESS) ? feats[(e0+m)*NBESS + k] : 0.0f);
    }
    wmma_layer<true , 32,  64, 32, 64 >(A1p, W1h, A2p, lane);
    wmma_layer<true , 64,  64, 64, 64 >(A2p, W2h, A3p, lane);
    wmma_layer<true , 64,  64, 64, 64 >(A3p, W3h, A2p, lane);
    wmma_layer<false, 64, 256, 64, 256>(A2p, W4h, Wp,  lane);

    for (int e = 0; e < 16; ++e) {
      const int ge = e0 + e;
      const int i  = idx_i[ge];
      const int s  = spec[idx_j[ge]];
      const float y0 = Y1[ge*3+0], y1v = Y1[ge*3+1], y2v = Y1[ge*3+2];
      for (int c = lane; c < H; c += 32) {
        const float hv = h0s[s*H + c];
        const float m0 = (float)Wp[e*256 +       c] * hv * INV_AVG;
        const float t  = (float)Wp[e*256 + 128 + c] * hv * INV_AVG;
        atomicAdd(&a0[i*H + c], m0);
        atomicAdd(&a1[(i*H + c)*3 + 0], t*y0);
        atomicAdd(&a1[(i*H + c)*3 + 1], t*y1v);
        atomicAdd(&a1[(i*H + c)*3 + 2], t*y2v);
      }
    }
  }
}

// ---------------------------------------------------------------------------
// Edge kernel 2: second radial-MLP (->5*H) + gathers of g0/g1 + cross product
// ---------------------------------------------------------------------------
__global__ void __launch_bounds__(32*EWAVES)
edge_mlp2_kernel(const float* __restrict__ feats, const float* __restrict__ Y1,
                 const int* __restrict__ idx_i, const int* __restrict__ idx_j,
                 const float* __restrict__ g0, const float* __restrict__ g1,
                 const float* __restrict__ R0, const float* __restrict__ R1,
                 const float* __restrict__ R2, const float* __restrict__ R3,
                 float* __restrict__ a0b, float* __restrict__ a1b) {
  __shared__ alignas(32) _Float16 W1h[32*64];
  __shared__ alignas(32) _Float16 W2h[64*64];
  __shared__ alignas(32) _Float16 W3h[64*64];
  __shared__ alignas(32) _Float16 W4h[64*640];      // 80 KB, fits 320 KB LDS
  __shared__ alignas(16) _Float16 act1[EWAVES][16*32];
  __shared__ alignas(16) _Float16 act2[EWAVES][16*64];
  __shared__ alignas(16) _Float16 act3[EWAVES][16*64];
  __shared__ alignas(16) _Float16 wt  [EWAVES][16*640];

  const int tid = threadIdx.x, nthr = blockDim.x;
  fill_packed<32,  64, NBESS>(W1h, R0, tid, nthr);
  fill_packed<64,  64, 64  >(W2h, R1, tid, nthr);
  fill_packed<64,  64, 64  >(W3h, R2, tid, nthr);
  fill_packed<64, 640, 64  >(W4h, R3, tid, nthr);
  __syncthreads();

  const int lane = tid & 31, wave = tid >> 5;
  const int ntiles = NEDGES / 16;
  for (int tile = blockIdx.x*EWAVES + wave; tile < ntiles; tile += gridDim.x*EWAVES) {
    const int e0 = tile * 16;
    _Float16* A1p = act1[wave];
    _Float16* A2p = act2[wave];
    _Float16* A3p = act3[wave];
    _Float16* Wp  = wt[wave];
    for (int idx = lane; idx < 16*32; idx += 32) {
      const int m = idx >> 5, k = idx & 31;
      A1p[idx] = (_Float16)((k < NBESS) ? feats[(e0+m)*NBESS + k] : 0.0f);
    }
    wmma_layer<true , 32,  64, 32, 64 >(A1p, W1h, A2p, lane);
    wmma_layer<true , 64,  64, 64, 64 >(A2p, W2h, A3p, lane);
    wmma_layer<true , 64,  64, 64, 64 >(A3p, W3h, A2p, lane);
    wmma_layer<false, 64, 640, 64, 640>(A2p, W4h, Wp,  lane);

    for (int e = 0; e < 16; ++e) {
      const int ge = e0 + e;
      const int i  = idx_i[ge];
      const int j  = idx_j[ge];
      const float y0 = Y1[ge*3+0], y1v = Y1[ge*3+1], y2v = Y1[ge*3+2];
      for (int c = lane; c < H; c += 32) {
        const float g0v = g0[j*H + c];
        const float gx = g1[(j*H + c)*3 + 0];
        const float gy = g1[(j*H + c)*3 + 1];
        const float gz = g1[(j*H + c)*3 + 2];
        const float w0 = (float)Wp[e*640 + 0*H + c];
        const float w1 = (float)Wp[e*640 + 1*H + c];
        const float w2 = (float)Wp[e*640 + 2*H + c];
        const float w3 = (float)Wp[e*640 + 3*H + c];
        const float w4 = (float)Wp[e*640 + 4*H + c];
        const float dotc = gx*y0 + gy*y1v + gz*y2v;
        const float cx = gy*y2v - gz*y1v;
        const float cy = gz*y0  - gx*y2v;
        const float cz = gx*y1v - gy*y0;
        const float b  = w1 * g0v;
        atomicAdd(&a0b[i*H + c], (w0*g0v + w3*dotc) * INV_AVG);
        atomicAdd(&a1b[(i*H + c)*3 + 0], (b*y0  + w2*gx + w4*cx) * INV_AVG);
        atomicAdd(&a1b[(i*H + c)*3 + 1], (b*y1v + w2*gy + w4*cy) * INV_AVG);
        atomicAdd(&a1b[(i*H + c)*3 + 2], (b*y2v + w2*gz + w4*cz) * INV_AVG);
      }
    }
  }
}

// ---------------------------------------------------------------------------
// Node kernel 1: A0/A1 projections, polynomial B0/B1, out0(+sc0), out1,
// sc0b, g0, g1. One block of H threads per node.
// ---------------------------------------------------------------------------
__global__ void __launch_bounds__(H)
node1_kernel(const float* __restrict__ a0, const float* __restrict__ a1,
             const int* __restrict__ spec,
             const float* __restrict__ Wout0, const float* __restrict__ Wout1,
             const float* __restrict__ Wp0,   const float* __restrict__ Wp1,
             const float* __restrict__ Wl0,   const float* __restrict__ Wl1,
             const float* __restrict__ sc0tab,
             const float* __restrict__ Wsc2,  const float* __restrict__ Wup0_2,
             const float* __restrict__ Wup1_2,
             float* __restrict__ out0_g, float* __restrict__ out1_g,
             float* __restrict__ sc0b_g, float* __restrict__ g0_g,
             float* __restrict__ g1_g) {
  __shared__ float a0s[H], a1s[H*3];
  __shared__ float B0s[PDIM], B1s[PDIM*3];
  __shared__ float out0s[H], out1s[H*3];
  const int n = blockIdx.x, t = threadIdx.x;
  const int s = spec[n];
  a0s[t] = a0[n*H + t];
  a1s[t*3+0] = a1[(n*H + t)*3 + 0];
  a1s[t*3+1] = a1[(n*H + t)*3 + 1];
  a1s[t*3+2] = a1[(n*H + t)*3 + 2];
  __syncthreads();

  if (t < PDIM) {
    float A0 = 0.f, Ax = 0.f, Ay = 0.f, Az = 0.f;
    for (int h = 0; h < H; ++h) {
      const float w0 = Wout0[h*PDIM + t];
      const float w1 = Wout1[h*PDIM + t];
      A0 = fmaf(a0s[h], w0, A0);
      Ax = fmaf(a1s[h*3+0], w1, Ax);
      Ay = fmaf(a1s[h*3+1], w1, Ay);
      Az = fmaf(a1s[h*3+2], w1, Az);
    }
    const float dotp = Ax*Ax + Ay*Ay + Az*Az;
    const float A02  = A0*A0;
    const float* wp0 = Wp0 + s*5*PDIM;
    const float* wp1 = Wp1 + s*4*PDIM;
    B0s[t] = wp0[0*PDIM+t]*A0 + wp0[1*PDIM+t]*A02 + wp0[2*PDIM+t]*A02*A0
           + wp0[3*PDIM+t]*dotp + wp0[4*PDIM+t]*A0*dotp;
    const float b1 = wp1[0*PDIM+t] + wp1[1*PDIM+t]*A0 + wp1[2*PDIM+t]*A02
                   + wp1[3*PDIM+t]*dotp;
    B1s[t*3+0] = b1*Ax; B1s[t*3+1] = b1*Ay; B1s[t*3+2] = b1*Az;
  }
  __syncthreads();

  {
    float o0 = sc0tab[s*H + t];
    float ox = 0.f, oy = 0.f, oz = 0.f;
    for (int p = 0; p < PDIM; ++p) {
      const float wl0 = Wl0[p*H + t];
      const float wl1 = Wl1[p*H + t];
      o0 = fmaf(B0s[p], wl0, o0);
      ox = fmaf(B1s[p*3+0], wl1, ox);
      oy = fmaf(B1s[p*3+1], wl1, oy);
      oz = fmaf(B1s[p*3+2], wl1, oz);
    }
    out0s[t] = o0;
    out1s[t*3+0] = ox; out1s[t*3+1] = oy; out1s[t*3+2] = oz;
    out0_g[n*H + t] = o0;                     // first output slice
    out1_g[(n*H + t)*3 + 0] = ox;
    out1_g[(n*H + t)*3 + 1] = oy;
    out1_g[(n*H + t)*3 + 2] = oz;
  }
  __syncthreads();

  {
    float sc = 0.f, g0v = 0.f, gx = 0.f, gy = 0.f, gz = 0.f;
    for (int c = 0; c < H; ++c) {
      const float o0c = out0s[c];
      sc  = fmaf(Wsc2[(s*H + c)*H + t], o0c, sc);
      g0v = fmaf(o0c, Wup0_2[c*H + t], g0v);
      const float wu1 = Wup1_2[c*H + t];
      gx = fmaf(out1s[c*3+0], wu1, gx);
      gy = fmaf(out1s[c*3+1], wu1, gy);
      gz = fmaf(out1s[c*3+2], wu1, gz);
    }
    sc0b_g[n*H + t] = sc;
    g0_g[n*H + t]   = g0v;
    g1_g[(n*H + t)*3 + 0] = gx;
    g1_g[(n*H + t)*3 + 1] = gy;
    g1_g[(n*H + t)*3 + 2] = gz;
  }
}

// ---------------------------------------------------------------------------
// Node kernel 2: final readout  out0b = _B0(wp0, A0, A1) @ Wl0_2 + sc0b
// ---------------------------------------------------------------------------
__global__ void __launch_bounds__(H)
node2_kernel(const float* __restrict__ a0b, const float* __restrict__ a1b,
             const int* __restrict__ spec,
             const float* __restrict__ Wout0, const float* __restrict__ Wout1,
             const float* __restrict__ Wp0,   const float* __restrict__ Wl0,
             const float* __restrict__ sc0b_g, float* __restrict__ outb) {
  __shared__ float a0s[H], a1s[H*3], B0s[PDIM];
  const int n = blockIdx.x, t = threadIdx.x;
  const int s = spec[n];
  a0s[t] = a0b[n*H + t];
  a1s[t*3+0] = a1b[(n*H + t)*3 + 0];
  a1s[t*3+1] = a1b[(n*H + t)*3 + 1];
  a1s[t*3+2] = a1b[(n*H + t)*3 + 2];
  __syncthreads();

  if (t < PDIM) {
    float A0 = 0.f, Ax = 0.f, Ay = 0.f, Az = 0.f;
    for (int h = 0; h < H; ++h) {
      const float w0 = Wout0[h*PDIM + t];
      const float w1 = Wout1[h*PDIM + t];
      A0 = fmaf(a0s[h], w0, A0);
      Ax = fmaf(a1s[h*3+0], w1, Ax);
      Ay = fmaf(a1s[h*3+1], w1, Ay);
      Az = fmaf(a1s[h*3+2], w1, Az);
    }
    const float dotp = Ax*Ax + Ay*Ay + Az*Az;
    const float A02  = A0*A0;
    const float* wp0 = Wp0 + s*5*PDIM;
    B0s[t] = wp0[0*PDIM+t]*A0 + wp0[1*PDIM+t]*A02 + wp0[2*PDIM+t]*A02*A0
           + wp0[3*PDIM+t]*dotp + wp0[4*PDIM+t]*A0*dotp;
  }
  __syncthreads();

  float o = sc0b_g[n*H + t];
  for (int p = 0; p < PDIM; ++p)
    o = fmaf(B0s[p], Wl0[p*H + t], o);
  outb[n*H + t] = o;   // second output slice
}

// ---------------------------------------------------------------------------
// Host-side launcher
// ---------------------------------------------------------------------------
extern "C" void kernel_launch(void* const* d_in, const int* in_sizes, int n_in,
                              void* d_out, int out_size, void* d_ws, size_t ws_size,
                              hipStream_t stream) {
  const float* positions = (const float*)d_in[0];
  const float* shifts    = (const float*)d_in[1];
  const float* node_attrs= (const float*)d_in[2];
  const float* W_embed   = (const float*)d_in[3];
  const float* Wsc1      = (const float*)d_in[4];
  const float* Wup0_1    = (const float*)d_in[5];
  const float* R1_0      = (const float*)d_in[6];
  const float* R1_1      = (const float*)d_in[7];
  const float* R1_2      = (const float*)d_in[8];
  const float* R1_3      = (const float*)d_in[9];
  const float* Wout0_1   = (const float*)d_in[10];
  const float* Wout1_1   = (const float*)d_in[11];
  const float* Wp0_1     = (const float*)d_in[12];
  const float* Wp1_1     = (const float*)d_in[13];
  const float* Wl0_1     = (const float*)d_in[14];
  const float* Wl1_1     = (const float*)d_in[15];
  const float* Wsc2      = (const float*)d_in[16];
  const float* Wup0_2    = (const float*)d_in[17];
  const float* Wup1_2    = (const float*)d_in[18];
  const float* R2_0      = (const float*)d_in[19];
  const float* R2_1      = (const float*)d_in[20];
  const float* R2_2      = (const float*)d_in[21];
  const float* R2_3      = (const float*)d_in[22];
  const float* Wout0_2   = (const float*)d_in[23];
  const float* Wout1_2   = (const float*)d_in[24];
  const float* Wp0_2     = (const float*)d_in[25];
  const float* Wl0_2     = (const float*)d_in[26];
  const int*   idx_i     = (const int*)d_in[27];
  const int*   idx_j     = (const int*)d_in[28];

  float* out0  = (float*)d_out;                 // [NNODES*H]
  float* out0b = (float*)d_out + NNODES*H;      // [NNODES*H]

  // workspace layout (floats); accumulators first so one zero pass covers them
  float* ws     = (float*)d_ws;
  float* A0acc  = ws;                            // NNODES*H
  float* A1acc  = A0acc  + NNODES*H;             // NNODES*H*3
  float* A0Bacc = A1acc  + NNODES*H*3;           // NNODES*H
  float* A1Bacc = A0Bacc + NNODES*H;             // NNODES*H*3
  float* Y1     = A1Bacc + NNODES*H*3;           // NEDGES*3
  float* feats  = Y1     + NEDGES*3;             // NEDGES*NBESS
  float* h0tab  = feats  + NEDGES*NBESS;         // NSPEC*H
  float* sc0tab = h0tab  + NSPEC*H;              // NSPEC*H
  float* out1w  = sc0tab + NSPEC*H;              // NNODES*H*3
  float* sc0bw  = out1w  + NNODES*H*3;           // NNODES*H
  float* g0w    = sc0bw  + NNODES*H;             // NNODES*H
  float* g1w    = g0w    + NNODES*H;             // NNODES*H*3
  int*   spec   = (int*)(g1w + NNODES*H*3);      // NNODES

  const int accN = NNODES*H*8;                   // a0+a1+a0b+a1b
  zero_kernel<<<2048, 256, 0, stream>>>(A0acc, accN);
  species_kernel<<<(NNODES+255)/256, 256, 0, stream>>>(node_attrs, spec);
  tables_kernel<<<NSPEC, H, 0, stream>>>(W_embed, Wup0_1, Wsc1, h0tab, sc0tab);
  edge_geom_kernel<<<(NEDGES+255)/256, 256, 0, stream>>>(positions, shifts, idx_i, idx_j,
                                                         Y1, feats);
  edge_mlp1_kernel<<<1000, 32*EWAVES, 0, stream>>>(feats, Y1, idx_i, idx_j, spec, h0tab,
                                                   R1_0, R1_1, R1_2, R1_3, A0acc, A1acc);
  node1_kernel<<<NNODES, H, 0, stream>>>(A0acc, A1acc, spec,
                                         Wout0_1, Wout1_1, Wp0_1, Wp1_1, Wl0_1, Wl1_1,
                                         sc0tab, Wsc2, Wup0_2, Wup1_2,
                                         out0, out1w, sc0bw, g0w, g1w);
  edge_mlp2_kernel<<<1000, 32*EWAVES, 0, stream>>>(feats, Y1, idx_i, idx_j, g0w, g1w,
                                                   R2_0, R2_1, R2_2, R2_3, A0Bacc, A1Bacc);
  node2_kernel<<<NNODES, H, 0, stream>>>(A0Bacc, A1Bacc, spec,
                                         Wout0_2, Wout1_2, Wp0_2, Wl0_2, sc0bw, out0b);
}
